// OptimizedGAT_23433341567027
// MI455X (gfx1250) — matrix operand order repaired and checked by reading, not compiled, not run on hardware
//
#include <hip/hip_runtime.h>
#include <hip/hip_bf16.h>

// ============================================================================
// GAT + link-prediction forward for MI455X (gfx1250, wave32, WMMA).
// Dense GEMMs use v_wmma_f32_16x16x32_f16 (f16 in, f32 accumulate); irregular
// graph ops use f32 global atomics. All kernels launch on `stream`.
// ============================================================================

typedef _Float16 half_t;
typedef __attribute__((ext_vector_type(16))) _Float16 v16h;
typedef __attribute__((ext_vector_type(8)))  _Float16 v8h;
typedef __attribute__((ext_vector_type(8)))  float    v8f;

static inline int cdiv(long long a, long long b) { return (int)((a + b - 1) / b); }

// ---------------------------------------------------------------- fills/convert
__global__ void k_fill_f32(float* p, float v, long long n) {
  long long i = blockIdx.x * (long long)blockDim.x + threadIdx.x;
  if (i < n) p[i] = v;
}
__global__ void k_fill_u32(unsigned* p, unsigned v, long long n) {
  long long i = blockIdx.x * (long long)blockDim.x + threadIdx.x;
  if (i < n) p[i] = v;
}
// Convert f32 weight to f16, optionally transposing (input stored as [N,K],
// output stored row-major [K,N]).
__global__ void k_cvt_f16(const float* __restrict__ W, half_t* __restrict__ O,
                          int K, int N, int trans) {
  long long i = blockIdx.x * (long long)blockDim.x + threadIdx.x;
  if (i >= (long long)K * N) return;
  int k = (int)(i / N), n = (int)(i % N);
  float v = trans ? W[(size_t)n * K + k] : W[i];
  O[i] = (half_t)v;
}
__global__ void k_bias_init(float* __restrict__ out, const float* __restrict__ b,
                            long long rows, int C) {
  long long i = blockIdx.x * (long long)blockDim.x + threadIdx.x;
  if (i >= rows * C) return;
  out[i] = b[i % C];
}

// ---------------------------------------------------------------- WMMA GEMM
// C[M,N] = act(A[M,K] @ Bh[K,N] + bias).  A: f32 (converted to f16 on LDS
// stage).  Bh: pre-converted f16, row-major K x N.  K % 32 == 0, N % 8 == 0.
// Block: 256 threads = 8 waves; block tile 128x128; wave tile 32x64
// (2x4 accumulators -> 8 WMMAs per 32-K step).
#define GEMM_TM 128
#define GEMM_TN 128
#define GEMM_TK 32

__launch_bounds__(256)
__global__ void k_gemm_wmma(const float* __restrict__ A, const half_t* __restrict__ Bh,
                            const float* __restrict__ bias, float* __restrict__ C,
                            int M, int N, int K, int act) {
  __shared__ __align__(16) _Float16 lsa[GEMM_TM][GEMM_TK + 8];  // 128 x 40 halves
  __shared__ __align__(16) _Float16 lsb[GEMM_TN][GEMM_TK + 8];  // [n][k] transposed

  const int tid  = threadIdx.x;
  const int lane = tid & 31;
  const int w    = tid >> 5;
  const int wr   = w & 3;          // wave row 0..3  (32 rows each)
  const int wc   = w >> 2;         // wave col 0..1  (64 cols each)
  const int hf   = lane >> 4;      // half-wave select
  const int l16  = lane & 15;

  const int row0 = blockIdx.y * GEMM_TM;
  const int col0 = blockIdx.x * GEMM_TN;
  const bool mfull = (row0 + GEMM_TM) <= M;   // block-uniform fast paths
  const bool nfull = (col0 + GEMM_TN) <= N;

  v8f acc[2][4] = {};

  for (int k0 = 0; k0 < K; k0 += GEMM_TK) {
    // ---- A tile: 128 rows x 32 f32 -> f16.  1024 float4 / 256 thr = 4 each.
    if (mfull) {
#pragma unroll
      for (int i = 0; i < 4; ++i) {
        int idx = tid + i * 256;
        int r = idx >> 3, seg = idx & 7;
        const float* ap = A + (size_t)(row0 + r) * K + k0 + seg * 4;
        float4 v = *(const float4*)ap;
        if (k0 + GEMM_TK < K) __builtin_prefetch(ap + GEMM_TK, 0, 1);
        _Float16* q = &lsa[r][seg * 4];
        q[0] = (_Float16)v.x; q[1] = (_Float16)v.y;
        q[2] = (_Float16)v.z; q[3] = (_Float16)v.w;
      }
    } else {
#pragma unroll
      for (int i = 0; i < 4; ++i) {
        int idx = tid + i * 256;
        int r = idx >> 3, seg = idx & 7;
        int gr = row0 + r;
        float4 v = make_float4(0.f, 0.f, 0.f, 0.f);
        if (gr < M) v = *(const float4*)(A + (size_t)gr * K + k0 + seg * 4);
        _Float16* q = &lsa[r][seg * 4];
        q[0] = (_Float16)v.x; q[1] = (_Float16)v.y;
        q[2] = (_Float16)v.z; q[3] = (_Float16)v.w;
      }
    }
    // ---- B tile: 32 rows (k) x 128 cols (n) f16, stored transposed in LDS.
    // 512 col-octets / 256 thr = 2 each.  gk always < K (K % 32 == 0).
#pragma unroll
    for (int t = 0; t < 2; ++t) {
      int idx = tid + t * 256;
      int r = idx >> 4, seg = idx & 15;
      int gk = k0 + r;
      int gn = col0 + seg * 8;
      const half_t* bp = Bh + (size_t)gk * N + gn;
      if (nfull) {
#pragma unroll
        for (int q = 0; q < 8; ++q) lsb[seg * 8 + q][r] = bp[q];
      } else {
#pragma unroll
        for (int q = 0; q < 8; ++q)
          lsb[seg * 8 + q][r] = (gn + q < N) ? bp[q] : (half_t)0.f;
      }
    }
    __syncthreads();

    // ---- fragments (layouts per CDNA5 ISA 7.12.2, wave32)
    v16h af[2], bf[4];
#pragma unroll
    for (int i = 0; i < 2; ++i) {
      const _Float16* pa = &lsa[wr * 32 + i * 16 + l16][hf * 8];
      v8h lo = *(const v8h*)pa;            // K = hf*8 + 0..7
      v8h hi = *(const v8h*)(pa + 16);     // K = 16 + hf*8 + 0..7
      af[i] = __builtin_shufflevector(lo, hi, 0,1,2,3,4,5,6,7,8,9,10,11,12,13,14,15);
    }
#pragma unroll
    for (int j = 0; j < 4; ++j) {
      const _Float16* pb = &lsb[wc * 64 + j * 16 + l16][hf * 8];
      v8h lo = *(const v8h*)pb;            // K = hf*8 + 0..7
      v8h hi = *(const v8h*)(pb + 16);     // K = 16 + hf*8 + 0..7
      bf[j] = __builtin_shufflevector(lo, hi, 0,1,2,3,4,5,6,7,8,9,10,11,12,13,14,15);
    }
#pragma unroll
    for (int i = 0; i < 2; ++i)
#pragma unroll
      for (int j = 0; j < 4; ++j)
        acc[i][j] = __builtin_amdgcn_wmma_f32_16x16x32_f16(
            false, af[i], false, bf[j], (short)0, acc[i][j], false, false);
    __syncthreads();
  }

  // ---- epilogue: bias + optional exact GELU
#pragma unroll
  for (int i = 0; i < 2; ++i)
#pragma unroll
    for (int j = 0; j < 4; ++j) {
      int n = col0 + wc * 64 + j * 16 + l16;
      if (n >= N) continue;
      float bv = bias ? bias[n] : 0.f;
#pragma unroll
      for (int r = 0; r < 8; ++r) {
        int m = row0 + wr * 32 + i * 16 + hf * 8 + r;
        if (m >= M) continue;
        float v = acc[i][j][r] + bv;
        if (act == 1) v = 0.5f * v * (1.f + erff(v * 0.70710678118654752f));
        C[(size_t)m * N + n] = v;
      }
    }
}

// ---------------------------------------------------------------- GAT pieces
// per-(node,head) score dots: ssrc = h . a_s, sdst = h . a_d  (one wave each)
__global__ void k_gat_scores(const float* __restrict__ h, const float* __restrict__ a_s,
                             const float* __restrict__ a_d, float* __restrict__ ssrc,
                             float* __restrict__ sdst, int N, int H, int C) {
  int gw = (int)((blockIdx.x * (long long)blockDim.x + threadIdx.x) >> 5);
  int lane = threadIdx.x & 31;
  if (gw >= N * H) return;
  int n = gw / H, hh = gw % H;
  const float* hp = h + (size_t)n * H * C + (size_t)hh * C;
  const float* as = a_s + (size_t)hh * C;
  const float* ad = a_d + (size_t)hh * C;
  float s1 = 0.f, s2 = 0.f;
  for (int c = lane; c < C; c += 32) { float v = hp[c]; s1 += v * as[c]; s2 += v * ad[c]; }
  for (int off = 16; off; off >>= 1) { s1 += __shfl_down(s1, off, 32); s2 += __shfl_down(s2, off, 32); }
  if (!lane) { ssrc[gw] = s1; sdst[gw] = s2; }
}

__device__ __forceinline__ unsigned f32_ordkey(float f) {
  unsigned u = __float_as_uint(f);
  return (u & 0x80000000u) ? ~u : (u | 0x80000000u);
}
__device__ __forceinline__ float f32_ordkey_inv(unsigned u) {
  return (u & 0x80000000u) ? __uint_as_float(u ^ 0x80000000u) : __uint_as_float(~u);
}

// per-(edge,head): e = leaky_relu(ssrc[src] + sdst[dst]); atomic segment max
__global__ void k_edge_max(const int* __restrict__ src, const int* __restrict__ dst,
                           const float* __restrict__ ssrc, const float* __restrict__ sdst,
                           float* __restrict__ ev, unsigned* __restrict__ mkey,
                           int Etot, int E, int H) {
  long long g = blockIdx.x * (long long)blockDim.x + threadIdx.x;
  if (g >= (long long)Etot * H) return;
  int e = (int)(g / H), hh = (int)(g % H);
  int s = (e < E) ? src[e] : (e - E);
  int d = (e < E) ? dst[e] : (e - E);
  float v = ssrc[(size_t)s * H + hh] + sdst[(size_t)d * H + hh];
  v = (v > 0.f) ? v : 0.2f * v;   // leaky_relu, slope 0.2
  ev[g] = v;
  atomicMax(&mkey[(size_t)d * H + hh], f32_ordkey(v));
}

// per-(edge,head): ex = exp(e - max[dst]); atomic segment sum
__global__ void k_edge_expsum(const int* __restrict__ dst, float* __restrict__ ev,
                              const unsigned* __restrict__ mkey, float* __restrict__ ssum,
                              int Etot, int E, int H) {
  long long g = blockIdx.x * (long long)blockDim.x + threadIdx.x;
  if (g >= (long long)Etot * H) return;
  int e = (int)(g / H), hh = (int)(g % H);
  int d = (e < E) ? dst[e] : (e - E);
  float m  = f32_ordkey_inv(mkey[(size_t)d * H + hh]);
  float ex = __expf(ev[g] - m);
  ev[g] = ex;
  atomicAdd(&ssum[(size_t)d * H + hh], ex);
}

// per-(edge,head) wave: out[dst] += alpha * h[src]  (lanes stride channels)
__global__ void k_gat_agg(const int* __restrict__ src, const int* __restrict__ dst,
                          const float* __restrict__ ev, const float* __restrict__ ssum,
                          const float* __restrict__ h, float* __restrict__ out,
                          int Etot, int E, int H, int C) {
  long long gw = (blockIdx.x * (long long)blockDim.x + threadIdx.x) >> 5;
  int lane = threadIdx.x & 31;
  if (gw >= (long long)Etot * H) return;
  int e = (int)(gw / H), hh = (int)(gw % H);
  int s = (e < E) ? src[e] : (e - E);
  int d = (e < E) ? dst[e] : (e - E);
  float alpha = ev[gw] / (ssum[(size_t)d * H + hh] + 1e-16f);
  const float* hp = h + ((size_t)s * H + hh) * C;
  float* op = out + ((size_t)d * H + hh) * C;
  for (int c = lane; c < C; c += 32) atomicAdd(&op[c], alpha * hp[c]);
}

// ---------------------------------------------------------------- BatchNorm
__global__ void k_bn_partial(const float* __restrict__ x, int Rows, int C,
                             float* __restrict__ sum, float* __restrict__ sq) {
  int c = blockIdx.x * blockDim.x + threadIdx.x;
  if (c >= C) return;
  float s = 0.f, q = 0.f;
  for (int r = blockIdx.y; r < Rows; r += gridDim.y) {
    float v = x[(size_t)r * C + c];
    s += v; q += v * v;
  }
  atomicAdd(&sum[c], s);
  atomicAdd(&sq[c], q);
}
// y = bn(x); optional ELU; optional residual add. In-place safe (elementwise).
__global__ void k_bn_apply(const float* __restrict__ xin, float* __restrict__ xout,
                           const float* __restrict__ resid, const float* __restrict__ sum,
                           const float* __restrict__ sq, const float* __restrict__ g,
                           const float* __restrict__ b, long long Rows, int C, int elu) {
  long long i = blockIdx.x * (long long)blockDim.x + threadIdx.x;
  if (i >= Rows * C) return;
  int c = (int)(i % C);
  float inv = 1.f / (float)Rows;
  float m = sum[c] * inv;
  float v = sq[c] * inv - m * m;
  float y = (xin[i] - m) * rsqrtf(v + 1e-5f) * g[c] + b[c];
  if (elu) y = (y > 0.f) ? y : expm1f(y);
  if (resid) y += resid[i];
  xout[i] = y;
}

// ---------------------------------------------------------------- head pieces
__global__ void k_gather_pairs(const float* __restrict__ x5, const int* __restrict__ ps,
                               const int* __restrict__ pd, float* __restrict__ pairs,
                               int chunk, int base) {
  long long i = blockIdx.x * (long long)blockDim.x + threadIdx.x;
  if (i >= (long long)chunk * 2 * 384) return;
  int row = (int)(i / 384), d = (int)(i % 384);
  int e = base + (row >> 1);
  int node = (row & 1) ? pd[e] : ps[e];
  pairs[i] = x5[(size_t)node * 384 + d];
}

// one wave per (edge,head): full 2-token attention; writes mean-over-tokens.
__global__ void k_attn(const float* __restrict__ qkv, float* __restrict__ obar,
                       int nEdges, int base) {
  const int H = 4, DH = 96, D = 384, D3 = 1152;
  long long gw = (blockIdx.x * (long long)blockDim.x + threadIdx.x) >> 5;
  int lane = threadIdx.x & 31;
  if (gw >= (long long)nEdges * H) return;
  int e = (int)(gw / H), hh = (int)(gw % H);
  const float* r0 = qkv + (size_t)(2 * e) * D3;
  const float* r1 = qkv + (size_t)(2 * e + 1) * D3;
  float q0[3], q1[3], k0[3], k1[3], v0[3], v1[3];
#pragma unroll
  for (int i = 0; i < 3; ++i) {
    int d = hh * DH + lane + i * 32;
    q0[i] = r0[d];         q1[i] = r1[d];
    k0[i] = r0[D + d];     k1[i] = r1[D + d];
    v0[i] = r0[2 * D + d]; v1[i] = r1[2 * D + d];
  }
  float s00 = 0.f, s01 = 0.f, s10 = 0.f, s11 = 0.f;
#pragma unroll
  for (int i = 0; i < 3; ++i) {
    s00 += q0[i] * k0[i]; s01 += q0[i] * k1[i];
    s10 += q1[i] * k0[i]; s11 += q1[i] * k1[i];
  }
  for (int off = 16; off; off >>= 1) {
    s00 += __shfl_down(s00, off, 32); s01 += __shfl_down(s01, off, 32);
    s10 += __shfl_down(s10, off, 32); s11 += __shfl_down(s11, off, 32);
  }
  s00 = __shfl(s00, 0, 32); s01 = __shfl(s01, 0, 32);
  s10 = __shfl(s10, 0, 32); s11 = __shfl(s11, 0, 32);
  const float sc = 0.10206207261596575f;  // 1/sqrt(96)
  s00 *= sc; s01 *= sc; s10 *= sc; s11 *= sc;
  float m0 = fmaxf(s00, s01), m1 = fmaxf(s10, s11);
  float e00 = __expf(s00 - m0), e01 = __expf(s01 - m0);
  float e10 = __expf(s10 - m1), e11 = __expf(s11 - m1);
  float a00 = e00 / (e00 + e01), a01 = e01 / (e00 + e01);
  float a10 = e10 / (e10 + e11), a11 = e11 / (e10 + e11);
#pragma unroll
  for (int i = 0; i < 3; ++i) {
    int d = hh * DH + lane + i * 32;
    float o = 0.5f * ((a00 + a10) * v0[i] + (a01 + a11) * v1[i]);
    obar[(size_t)(base + e) * D + d] = o;
  }
}

__global__ void k_build_z(const float* __restrict__ x5, const int* __restrict__ ps,
                          const int* __restrict__ pd, const float* __restrict__ feat,
                          float* __restrict__ z, int Ep) {
  long long i = blockIdx.x * (long long)blockDim.x + threadIdx.x;
  if (i >= (long long)Ep * 1152) return;
  int row = (int)(i / 1152), d = (int)(i % 1152);
  float v;
  if (d < 384)      v = x5[(size_t)ps[row] * 384 + d];
  else if (d < 768) v = x5[(size_t)pd[row] * 384 + (d - 384)];
  else              v = feat[(size_t)row * 384 + (d - 768)];
  z[i] = v;
}

// final 96-dot + sigmoid (one wave per row)
__global__ void k_final(const float* __restrict__ g4, const float* __restrict__ w5,
                        const float* __restrict__ b5, float* __restrict__ out, int M) {
  long long gw = (blockIdx.x * (long long)blockDim.x + threadIdx.x) >> 5;
  int lane = threadIdx.x & 31;
  if (gw >= M) return;
  const float* p = g4 + (size_t)gw * 96;
  float s = 0.f;
  for (int i = lane; i < 96; i += 32) s += p[i] * w5[i];
  for (int off = 16; off; off >>= 1) s += __shfl_down(s, off, 32);
  if (!lane) out[gw] = 1.f / (1.f + __expf(-(s + b5[0])));
}

// ============================================================================
extern "C" void kernel_launch(void* const* d_in, const int* in_sizes, int n_in,
                              void* d_out, int out_size, void* d_ws, size_t ws_size,
                              hipStream_t stream) {
  (void)in_sizes; (void)n_in; (void)out_size;
  const int N = 10000, Fin = 128, Hd = 768, Od = 384, E = 320000, Ep = 20000;
  const int Etot = E + N;
  const int CHUNK = 5000;  // head edge chunk (Ep % CHUNK == 0)

  // ---- inputs (setup_inputs dict order, params flattened in insertion order)
  const float* x    = (const float*)d_in[0];
  const int*   ei   = (const int*)d_in[1];
  const int*   pei  = (const int*)d_in[2];
  const float* cW[5]  = {(const float*)d_in[3],  (const float*)d_in[7],
                         (const float*)d_in[11], (const float*)d_in[15],
                         (const float*)d_in[19]};
  const float* cAS[5] = {(const float*)d_in[4],  (const float*)d_in[8],
                         (const float*)d_in[12], (const float*)d_in[16],
                         (const float*)d_in[20]};
  const float* cAD[5] = {(const float*)d_in[5],  (const float*)d_in[9],
                         (const float*)d_in[13], (const float*)d_in[17],
                         (const float*)d_in[21]};
  const float* cB[5]  = {(const float*)d_in[6],  (const float*)d_in[10],
                         (const float*)d_in[14], (const float*)d_in[18],
                         (const float*)d_in[22]};
  const float* bnG[5] = {(const float*)d_in[23], (const float*)d_in[25],
                         (const float*)d_in[27], (const float*)d_in[29],
                         (const float*)d_in[31]};
  const float* bnB[5] = {(const float*)d_in[24], (const float*)d_in[26],
                         (const float*)d_in[28], (const float*)d_in[30],
                         (const float*)d_in[32]};
  const float* res1w = (const float*)d_in[33]; const float* res1b = (const float*)d_in[34];
  const float* res4w = (const float*)d_in[35]; const float* res4b = (const float*)d_in[36];
  const float* wqkv  = (const float*)d_in[37]; const float* bqkv  = (const float*)d_in[38];
  const float* wo    = (const float*)d_in[39]; const float* bo    = (const float*)d_in[40];
  const float* lpW[5] = {(const float*)d_in[41], (const float*)d_in[43],
                         (const float*)d_in[45], (const float*)d_in[47],
                         (const float*)d_in[49]};
  const float* lpB[5] = {(const float*)d_in[42], (const float*)d_in[44],
                         (const float*)d_in[46], (const float*)d_in[48],
                         (const float*)d_in[50]};
  const float* lpbnG[3] = {(const float*)d_in[51], (const float*)d_in[53],
                           (const float*)d_in[55]};
  const float* lpbnB[3] = {(const float*)d_in[52], (const float*)d_in[54],
                           (const float*)d_in[56]};

  const int* srcp = ei;       const int* dstp = ei + E;
  const int* psrc = pei;      const int* pdst = pei + Ep;
  float* outp = (float*)d_out;

  // ---- workspace arena (hand-overlaid; phases do not alias live buffers)
  char* ws = (char*)d_ws;
  const size_t X1o = 0,          XAo = 30720000, XBo = 61440000, HBo = 92160000;
  const size_t X5o = 122880000;
  const size_t SSo = 138240000,  SDo = 138560000, MKo = 138880000, SMo = 139200000;
  const size_t EVo = 139520000;                     // 10,560,000 B, ends 150,080,000
  // phase-B overlays of arenas [0,153.6M):
  const size_t PAIRSo = 0, QKVo = 15360000, OBARo = 61440000;
  const size_t Zo = 0, G1o = 92160000, G2o = 0, G3o = 92160000, G4o = 0;
  // persistent tail
  size_t tail = 153600000;
  auto talloc = [&](size_t b) { size_t o = tail; tail = (tail + b + 255) & ~(size_t)255; return o; };
  const size_t SUMo = talloc(1152 * 4);
  const size_t SQo  = talloc(1152 * 4);
  const size_t FEATo = talloc((size_t)Ep * Od * 4);
  // f16 weights
  const size_t W1o = talloc((size_t)128 * 768 * 2);
  const size_t W2o = talloc((size_t)768 * 768 * 2);
  const size_t W3o = talloc((size_t)768 * 768 * 2);
  const size_t W4o = talloc((size_t)768 * 768 * 2);
  const size_t W5o = talloc((size_t)768 * 384 * 2);
  const size_t R1o = talloc((size_t)128 * 768 * 2);
  const size_t R4o = talloc((size_t)768 * 384 * 2);
  const size_t QWo = talloc((size_t)384 * 1152 * 2);
  const size_t WOo = talloc((size_t)384 * 384 * 2);
  const size_t L1o = talloc((size_t)1152 * 768 * 2);
  const size_t L2o = talloc((size_t)768 * 384 * 2);
  const size_t L3o = talloc((size_t)384 * 192 * 2);
  const size_t L4o = talloc((size_t)192 * 96 * 2);
  if (ws_size < tail) return;  // workspace too small; bail safely

  float* X1p = (float*)(ws + X1o); float* XAp = (float*)(ws + XAo);
  float* XBp = (float*)(ws + XBo); float* HBp = (float*)(ws + HBo);
  float* X5p = (float*)(ws + X5o);
  float* SSp = (float*)(ws + SSo); float* SDp = (float*)(ws + SDo);
  unsigned* MKp = (unsigned*)(ws + MKo); float* SMp = (float*)(ws + SMo);
  float* EVp = (float*)(ws + EVo);
  float* SUMp = (float*)(ws + SUMo); float* SQp = (float*)(ws + SQo);
  float* FEATp = (float*)(ws + FEATo);
  float* PAIRSp = (float*)(ws + PAIRSo); float* QKVp = (float*)(ws + QKVo);
  float* OBARp = (float*)(ws + OBARo);
  float* Zp = (float*)(ws + Zo); float* G1p = (float*)(ws + G1o);
  float* G2p = (float*)(ws + G2o); float* G3p = (float*)(ws + G3o);
  float* G4p = (float*)(ws + G4o);
  half_t* Wh[5] = {(half_t*)(ws + W1o), (half_t*)(ws + W2o), (half_t*)(ws + W3o),
                   (half_t*)(ws + W4o), (half_t*)(ws + W5o)};
  half_t* R1h = (half_t*)(ws + R1o); half_t* R4h = (half_t*)(ws + R4o);
  half_t* QWh = (half_t*)(ws + QWo); half_t* WOh = (half_t*)(ws + WOo);
  half_t* Lh[4] = {(half_t*)(ws + L1o), (half_t*)(ws + L2o),
                   (half_t*)(ws + L3o), (half_t*)(ws + L4o)};

  // ---- launch helpers
  auto gemm = [&](const float* A, const half_t* Bh, const float* bias, float* Cp,
                  int M, int Nn, int K, int act) {
    dim3 g(cdiv(Nn, GEMM_TN), cdiv(M, GEMM_TM));
    k_gemm_wmma<<<g, 256, 0, stream>>>(A, Bh, bias, Cp, M, Nn, K, act);
  };
  auto cvt = [&](const float* W, half_t* O, int K, int Nn, int trans) {
    k_cvt_f16<<<cdiv((long long)K * Nn, 256), 256, 0, stream>>>(W, O, K, Nn, trans);
  };
  auto bn = [&](float* xin, float* xout, const float* resid, const float* g,
                const float* b, long long Rows, int C, int elu) {
    k_fill_f32<<<cdiv(C, 256), 256, 0, stream>>>(SUMp, 0.f, C);
    k_fill_f32<<<cdiv(C, 256), 256, 0, stream>>>(SQp, 0.f, C);
    k_bn_partial<<<dim3(cdiv(C, 256), 64), 256, 0, stream>>>(xin, (int)Rows, C, SUMp, SQp);
    k_bn_apply<<<cdiv(Rows * C, 256), 256, 0, stream>>>(xin, xout, resid, SUMp, SQp,
                                                        g, b, Rows, C, elu);
  };
  auto gat = [&](const float* xin, int li, int Fi, int Hh, int Cc, float* outbuf) {
    int HC = Hh * Cc;
    gemm(xin, Wh[li], nullptr, HBp, N, HC, Fi, 0);                 // h = x @ W
    k_gat_scores<<<cdiv((long long)N * Hh * 32, 256), 256, 0, stream>>>(
        HBp, cAS[li], cAD[li], SSp, SDp, N, Hh, Cc);
    k_fill_u32<<<cdiv((long long)N * Hh, 256), 256, 0, stream>>>(MKp, 0u, (long long)N * Hh);
    k_fill_f32<<<cdiv((long long)N * Hh, 256), 256, 0, stream>>>(SMp, 0.f, (long long)N * Hh);
    k_edge_max<<<cdiv((long long)Etot * Hh, 256), 256, 0, stream>>>(
        srcp, dstp, SSp, SDp, EVp, MKp, Etot, E, Hh);
    k_edge_expsum<<<cdiv((long long)Etot * Hh, 256), 256, 0, stream>>>(
        dstp, EVp, MKp, SMp, Etot, E, Hh);
    k_bias_init<<<cdiv((long long)N * HC, 256), 256, 0, stream>>>(outbuf, cB[li], N, HC);
    k_gat_agg<<<cdiv((long long)Etot * Hh * 32, 256), 256, 0, stream>>>(
        srcp, dstp, EVp, SMp, HBp, outbuf, Etot, E, Hh, Cc);
  };

  // ---- weight conversion (f32 -> f16, transposing torch-style [out,in] mats)
  cvt(cW[0], Wh[0], Fin, Hd, 0);
  cvt(cW[1], Wh[1], Hd, Hd, 0);
  cvt(cW[2], Wh[2], Hd, Hd, 0);
  cvt(cW[3], Wh[3], Hd, Hd, 0);
  cvt(cW[4], Wh[4], Hd, Od, 0);
  cvt(res1w, R1h, Fin, Hd, 0);
  cvt(res4w, R4h, Hd, Od, 0);
  cvt(wqkv, QWh, Od, 3 * Od, 1);   // (1152,384) -> [384 x 1152]
  cvt(wo, WOh, Od, Od, 1);
  cvt(lpW[0], Lh[0], 1152, 768, 0);
  cvt(lpW[1], Lh[1], 768, 384, 0);
  cvt(lpW[2], Lh[2], 384, 192, 0);
  cvt(lpW[3], Lh[3], 192, 96, 0);

  // ---- graph phase -------------------------------------------------------
  gemm(x, R1h, res1b, X1p, N, Hd, Fin, 0);                 // identity1 -> X1
  gat(x, 0, Fin, 8, 96, XAp);                              // gat1 -> XA
  bn(XAp, X1p, X1p, bnG[0], bnB[0], N, Hd, 1);             // x1 = elu(bn)+id1 -> X1
  gat(X1p, 1, Hd, 8, 96, XBp);
  bn(XBp, XBp, X1p, bnG[1], bnB[1], N, Hd, 1);             // x2 -> XB
  gat(XBp, 2, Hd, 4, 192, XAp);
  bn(XAp, XAp, XBp, bnG[2], bnB[2], N, Hd, 1);             // x3 -> XA
  gat(XAp, 3, Hd, 4, 192, XBp);
  bn(XBp, XBp, XAp, bnG[3], bnB[3], N, Hd, 1);             // x4 -> XB
  gemm(X1p, R4h, res4b, X5p, N, Od, Hd, 0);                // identity4 -> X5
  gat(XBp, 4, Hd, 1, 384, XAp);                            // gat5 -> XA
  bn(XAp, X5p, X5p, bnG[4], bnB[4], N, Od, 0);             // x5 = bn + id4 -> X5

  // ---- head: MHA over (src,dst) pairs, mean folded before W_o ------------
  for (int ch = 0; ch < Ep / CHUNK; ++ch) {
    int base = ch * CHUNK;
    k_gather_pairs<<<cdiv((long long)CHUNK * 2 * 384, 256), 256, 0, stream>>>(
        X5p, psrc, pdst, PAIRSp, CHUNK, base);
    gemm(PAIRSp, QWh, bqkv, QKVp, 2 * CHUNK, 3 * Od, Od, 0);
    k_attn<<<cdiv((long long)CHUNK * 4 * 32, 256), 256, 0, stream>>>(QKVp, OBARp, CHUNK, base);
  }
  gemm(OBARp, WOh, bo, FEATp, Ep, Od, Od, 0);              // feat = obar @ wo^T + bo
  k_build_z<<<cdiv((long long)Ep * 1152, 256), 256, 0, stream>>>(
      X5p, psrc, pdst, FEATp, Zp, Ep);

  // ---- MLP head ----------------------------------------------------------
  gemm(Zp, Lh[0], lpB[0], G1p, Ep, 768, 1152, 1);          // gelu fused
  bn(G1p, G1p, nullptr, lpbnG[0], lpbnB[0], Ep, 768, 0);
  gemm(G1p, Lh[1], lpB[1], G2p, Ep, 384, 768, 1);
  bn(G2p, G2p, nullptr, lpbnG[1], lpbnB[1], Ep, 384, 0);
  gemm(G2p, Lh[2], lpB[2], G3p, Ep, 192, 384, 1);
  bn(G3p, G3p, nullptr, lpbnG[2], lpbnB[2], Ep, 192, 0);
  gemm(G3p, Lh[3], lpB[3], G4p, Ep, 96, 192, 1);
  k_final<<<cdiv((long long)Ep * 32, 256), 256, 0, stream>>>(G4p, lpW[4], lpB[4], outp, Ep);
}
